// TransformationSpaceProjection_74801150427408
// MI455X (gfx1250) — matrix operand order repaired and checked by reading, not compiled, not run on hardware
//
#include <hip/hip_runtime.h>
#include <math.h>

// ---------------------------------------------------------------------------
// Types for CDNA5 WMMA (wave32): v_wmma_f32_16x16x32_bf16
// ---------------------------------------------------------------------------
typedef __attribute__((ext_vector_type(16))) __bf16          v16bf;
typedef __attribute__((ext_vector_type(8)))  float           v8f;
typedef __attribute__((ext_vector_type(4)))  unsigned int    u32x4;
typedef __attribute__((ext_vector_type(4)))  unsigned short  u16x4;

union Frag {
    v16bf v;
    u32x4 q[2];
};
union U128 {
    u32x4 q;
    unsigned short s[8];
};

__device__ __forceinline__ unsigned short f2b(float f) {
    // round-to-nearest-even f32 -> bf16
    unsigned int u = __float_as_uint(f);
    u += 0x7FFFu + ((u >> 16) & 1u);
    return (unsigned short)(u >> 16);
}

__device__ __forceinline__ unsigned int pack2(unsigned short lo, unsigned short hi) {
    return (unsigned int)lo | ((unsigned int)hi << 16);
}

// ---------------------------------------------------------------------------
// bf16-operand WMMA GEMM:  C[M,N] = act( (op(A) @ op(B)) * scale + bias )
//   TA=0: A stored [M,K] bf16, lda = row stride.  TA=1: A stored [K,M] (A^T)
//   TB=0: B stored [K,N] bf16, ldb = row stride.  TB=1: B stored [N,K] (B^T)
//   OBF=0: C is f32.  OBF=1: C is bf16.
// Block tile 128x128, K-step 32, 256 threads = 8 waves, wave tile 64x32,
// double-buffered LDS (one barrier per K-step), pure-copy loaders.
// Transpose staging writes k-pairs as dwords (8x ds_store_b32 per thread).
// M,N multiples of 128; K multiple of 32 (true for this workload).
// ---------------------------------------------------------------------------
template <bool TA, bool TB, bool OBF>
__global__ __launch_bounds__(256) void gemm_bf16_kernel(
    const unsigned short* __restrict__ A, int lda,
    const unsigned short* __restrict__ B, int ldb,
    void* __restrict__ Cv, int ldc, int K,
    const float* __restrict__ bias, float scale, int relu)
{
    constexpr int KP = 40; // padded LDS row stride (elems): 80B -> conflict-free b128
    __shared__ __align__(16) unsigned short lsA[2][128 * KP]; // [m][k] bf16
    __shared__ __align__(16) unsigned short lsB[2][128 * KP]; // [n][k] bf16

    const int t    = threadIdx.x;
    const int lane = t & 31;
    const int w    = t >> 5;       // wave 0..7
    const int wr   = w >> 2;       // 0..1  -> 64-row strip
    const int wc   = w & 3;        // 0..3  -> 32-col strip
    const int hi   = lane >> 4;    // half-wave select
    const int lr   = lane & 15;

    const int m0 = blockIdx.y * 128;
    const int n0 = blockIdx.x * 128;

    // each thread stages 2x16B per operand per K-step
    auto load_regs = [&](int kn, U128& ra0, U128& ra1, U128& rb0, U128& rb1) {
        if (!TA) { // contiguous: mm = t>>2 (+64), kc = (t&3)*8
            const size_t base = (size_t)(m0 + (t >> 2)) * lda + kn + (t & 3) * 8;
            ra0.q = *(const u32x4*)(A + base);
            ra1.q = *(const u32x4*)(A + base + (size_t)64 * lda);
        } else {   // k-pair rows: kk = 2*(t>>4), 2*(t>>4)+1; mc = (t&15)*8
            const size_t base = (size_t)(kn + 2 * (t >> 4)) * lda + m0 + (t & 15) * 8;
            ra0.q = *(const u32x4*)(A + base);
            ra1.q = *(const u32x4*)(A + base + (size_t)lda);
        }
        if (!TB) { // k-pair rows: kk = 2*(t>>4), 2*(t>>4)+1; nc = (t&15)*8
            const size_t base = (size_t)(kn + 2 * (t >> 4)) * ldb + n0 + (t & 15) * 8;
            rb0.q = *(const u32x4*)(B + base);
            rb1.q = *(const u32x4*)(B + base + (size_t)ldb);
        } else {   // contiguous: nn = t>>2 (+64), kc = (t&3)*8
            const size_t base = (size_t)(n0 + (t >> 2)) * ldb + kn + (t & 3) * 8;
            rb0.q = *(const u32x4*)(B + base);
            rb1.q = *(const u32x4*)(B + base + (size_t)64 * ldb);
        }
    };
    auto store_lds = [&](int buf, const U128& ra0, const U128& ra1,
                         const U128& rb0, const U128& rb1) {
        unsigned short* a = lsA[buf];
        unsigned short* b = lsB[buf];
        if (!TA) {
            *(u32x4*)&a[(t >> 2) * KP + (t & 3) * 8]        = ra0.q;
            *(u32x4*)&a[((t >> 2) + 64) * KP + (t & 3) * 8] = ra1.q;
        } else {
            const int kk = 2 * (t >> 4); // even -> dword aligned
#pragma unroll
            for (int i = 0; i < 8; ++i)
                *(unsigned int*)&a[((t & 15) * 8 + i) * KP + kk] = pack2(ra0.s[i], ra1.s[i]);
        }
        if (!TB) {
            const int kk = 2 * (t >> 4);
#pragma unroll
            for (int i = 0; i < 8; ++i)
                *(unsigned int*)&b[((t & 15) * 8 + i) * KP + kk] = pack2(rb0.s[i], rb1.s[i]);
        } else {
            *(u32x4*)&b[(t >> 2) * KP + (t & 3) * 8]        = rb0.q;
            *(u32x4*)&b[((t >> 2) + 64) * KP + (t & 3) * 8] = rb1.q;
        }
    };

    v8f acc[4][2];
#pragma unroll
    for (int i = 0; i < 4; ++i)
#pragma unroll
        for (int j = 0; j < 2; ++j)
            acc[i][j] = v8f{0.f, 0.f, 0.f, 0.f, 0.f, 0.f, 0.f, 0.f};

    // prologue: stage tile kt=0 into buffer 0
    {
        U128 ra0, ra1, rb0, rb1;
        load_regs(0, ra0, ra1, rb0, rb1);
        store_lds(0, ra0, ra1, rb0, rb1);
    }
    __syncthreads();

    int cur = 0;
    for (int kt = 0; kt < K; kt += 32) {
        const bool more = (kt + 32 < K);
        U128 ra0, ra1, rb0, rb1;
        if (more) load_regs(kt + 32, ra0, ra1, rb0, rb1); // issue early, WMMA hides it

        // ---- fragments per documented CDNA5 VGPR layouts ----
        // A 16x32 bf16: lane(lr,hi): M=lr; e0..7 -> K=8*hi+0..7, e8..15 -> K=16+8*hi+0..7
        const unsigned short* la = lsA[cur];
        const unsigned short* lb = lsB[cur];
        Frag aF[4];
#pragma unroll
        for (int i = 0; i < 4; ++i) {
            const int m = wr * 64 + i * 16 + lr;
            aF[i].q[0] = *(const u32x4*)&la[m * KP + 8 * hi];
            aF[i].q[1] = *(const u32x4*)&la[m * KP + 16 + 8 * hi];
        }
        // B 32x16 bf16: lane(lr,hi): N=lr; e0..7 -> K=8*hi+0..7, e8..15 -> K=16+8*hi+0..7
        Frag bF[2];
#pragma unroll
        for (int j = 0; j < 2; ++j) {
            const int n = wc * 32 + j * 16 + lr;
            bF[j].q[0] = *(const u32x4*)&lb[n * KP + 8 * hi];
            bF[j].q[1] = *(const u32x4*)&lb[n * KP + 16 + 8 * hi];
        }

#pragma unroll
        for (int i = 0; i < 4; ++i)
#pragma unroll
            for (int j = 0; j < 2; ++j)
                acc[i][j] = __builtin_amdgcn_wmma_f32_16x16x32_bf16(
                    false, aF[i].v, false, bF[j].v, (short)0, acc[i][j], false, false);

        if (more) store_lds(cur ^ 1, ra0, ra1, rb0, rb1);
        __syncthreads();
        cur ^= 1;
    }

    // ---- epilogue: scale, bias, relu; store f32 or bf16 ----
    // C/D layout: VGPR r, lane(lr,hi): M = r + 8*hi, N = lr
#pragma unroll
    for (int i = 0; i < 4; ++i) {
#pragma unroll
        for (int j = 0; j < 2; ++j) {
            const int nn = n0 + wc * 32 + j * 16 + lr;
            const float bv = bias ? bias[nn] : 0.0f;
#pragma unroll
            for (int r = 0; r < 8; ++r) {
                const int mm = m0 + wr * 64 + i * 16 + r + 8 * hi;
                float v = acc[i][j][r] * scale + bv;
                if (relu) v = fmaxf(v, 0.0f);
                if (OBF) ((unsigned short*)Cv)[(size_t)mm * ldc + nn] = f2b(v);
                else     ((float*)Cv)[(size_t)mm * ldc + nn] = v;
            }
        }
    }
}

// ---------------------------------------------------------------------------
// Row softmax: reads f32 logits [row][0..2048), writes bf16. One block/row,
// 256 threads, 8 cols/thread held in registers (single global read pass).
// ---------------------------------------------------------------------------
__global__ __launch_bounds__(256) void softmax_rows_bf16_kernel(
    const float* __restrict__ in, unsigned short* __restrict__ outb)
{
    constexpr int NC = 2048;
    __shared__ float red[256];
    const int tid = threadIdx.x;
    const float* p = in + (size_t)blockIdx.x * NC;
    unsigned short* q = outb + (size_t)blockIdx.x * NC;

    float v[8];
    float lmax = -3.402823466e38f;
#pragma unroll
    for (int i = 0; i < 8; ++i) {
        v[i] = p[tid + i * 256];
        lmax = fmaxf(lmax, v[i]);
    }
    red[tid] = lmax;
    __syncthreads();
    for (int s = 128; s > 0; s >>= 1) {
        if (tid < s) red[tid] = fmaxf(red[tid], red[tid + s]);
        __syncthreads();
    }
    const float rmax = red[0];
    __syncthreads();

    float lsum = 0.0f;
#pragma unroll
    for (int i = 0; i < 8; ++i) {
        v[i] = __expf(v[i] - rmax);
        lsum += v[i];
    }
    red[tid] = lsum;
    __syncthreads();
    for (int s = 128; s > 0; s >>= 1) {
        if (tid < s) red[tid] += red[tid + s];
        __syncthreads();
    }
    const float inv = 1.0f / red[0];
#pragma unroll
    for (int i = 0; i < 8; ++i) q[tid + i * 256] = f2b(v[i] * inv);
}

// ---------------------------------------------------------------------------
// cat_bf16[s] = bf16([x[s] | y[s]])  (f32 in, bf16 out, float4 granularity)
// ---------------------------------------------------------------------------
__global__ __launch_bounds__(256) void concat_cvt_kernel(
    const float4* __restrict__ x, const float4* __restrict__ y,
    u16x4* __restrict__ out, int rows)
{
    const size_t g = (size_t)blockIdx.x * 256 + threadIdx.x;
    const size_t total = (size_t)rows * 512; // 512 float4 per 2048-wide out row
    if (g >= total) return;
    const size_t s = g >> 9;
    const int c = (int)(g & 511);
    const float4 f = (c < 256) ? x[s * 256 + c] : y[s * 256 + (c - 256)];
    out[g] = u16x4{f2b(f.x), f2b(f.y), f2b(f.z), f2b(f.w)};
}

// f32 -> bf16 elementwise
__global__ __launch_bounds__(256) void cvt_bf16_kernel(
    const float4* __restrict__ in, u16x4* __restrict__ out, size_t n4)
{
    const size_t g = (size_t)blockIdx.x * 256 + threadIdx.x;
    if (g >= n4) return;
    const float4 f = in[g];
    out[g] = u16x4{f2b(f.x), f2b(f.y), f2b(f.z), f2b(f.w)};
}

// ---------------------------------------------------------------------------
// Host-side launch helpers
// ---------------------------------------------------------------------------
typedef unsigned short bf16_t;

template <bool TA, bool TB, bool OBF>
static void run_gemm(const bf16_t* A, int lda, const bf16_t* B, int ldb,
                     void* C, int ldc, int M, int N, int K,
                     const float* bias, float scale, int relu, hipStream_t s)
{
    dim3 grid(N / 128, M / 128);
    hipLaunchKernelGGL((gemm_bf16_kernel<TA, TB, OBF>), grid, dim3(256), 0, s,
                       A, lda, B, ldb, C, ldc, K, bias, scale, relu);
}

static void run_cvt(const float* in, bf16_t* out, size_t n, hipStream_t s)
{
    const size_t n4 = n / 4;
    cvt_bf16_kernel<<<(unsigned)((n4 + 255) / 256), 256, 0, s>>>(
        (const float4*)in, (u16x4*)out, n4);
}

extern "C" void kernel_launch(void* const* d_in, const int* in_sizes, int n_in,
                              void* d_out, int out_size, void* d_ws, size_t ws_size,
                              hipStream_t stream)
{
    (void)in_sizes; (void)n_in; (void)out_size; (void)ws_size;

    const float* x     = (const float*)d_in[0];  // input_embedding   [8192,1024]
    const float* yemb  = (const float*)d_in[1];  // output_embedding  [8192,1024]
    const float* raug  = (const float*)d_in[2];  // random_augment    [8192,1024]
    const float* W_in  = (const float*)d_in[3];  // [1024,2048]
    const float* b_in  = (const float*)d_in[4];  // [2048]
    const float* W_out = (const float*)d_in[5];  // [1024,2048]
    const float* b_out = (const float*)d_in[6];  // [2048]
    const float* W_cat = (const float*)d_in[7];  // [2048,2048]
    const float* b_cat = (const float*)d_in[8];  // [2048]
    const float* W_fin = (const float*)d_in[9];  // [2048,1024]
    const float* b_fin = (const float*)d_in[10]; // [1024]
    float* out = (float*)d_out;                  // [2, 8192, 1024] flat

    const int S = 8192, D = 1024, D2 = 2048, OUTN = 1024;

    // workspace carve (256B aligned)
    char* p = (char*)d_ws;
    auto carveF = [&](size_t elems) -> float* {
        float* q = (float*)p;
        p += (elems * sizeof(float) + 255) & ~(size_t)255;
        return q;
    };
    auto carveH = [&](size_t elems) -> bf16_t* {
        bf16_t* q = (bf16_t*)p;
        p += (elems * sizeof(bf16_t) + 255) & ~(size_t)255;
        return q;
    };
    float*  lg    = carveF((size_t)S * D2);    // f32 logits          (64 MB)
    bf16_t* catb  = carveH((size_t)S * D2);    // cat bf16 / res bf16 (32 MB)
    bf16_t* icmb  = carveH((size_t)S * D2);    // softmax(cat@Wcat)   (32 MB)
    bf16_t* qb    = carveH((size_t)S * D2);    // iq / oq bf16        (32 MB)
    bf16_t* iqkb  = carveH((size_t)D2 * D2);   // ( 8 MB)
    bf16_t* oqkb  = carveH((size_t)D2 * D2);   // ( 8 MB)
    bf16_t* metab = carveH((size_t)D2 * D2);   // ( 8 MB)
    bf16_t* Winb  = carveH((size_t)D * D2);    // ( 4 MB)
    bf16_t* Woutb = carveH((size_t)D * D2);    // ( 4 MB)
    bf16_t* Wcatb = carveH((size_t)D2 * D2);   // ( 8 MB)
    bf16_t* Wfinb = carveH((size_t)D2 * OUTN); // ( 4 MB)

    // one-time weight conversion to bf16
    run_cvt(W_in,  Winb,  (size_t)D * D2,    stream);
    run_cvt(W_out, Woutb, (size_t)D * D2,    stream);
    run_cvt(W_cat, Wcatb, (size_t)D2 * D2,   stream);
    run_cvt(W_fin, Wfinb, (size_t)D2 * OUTN, stream);

    const float inv_sqrt = 1.0f / sqrtf((float)D2);

    auto march = [&](const float* xa, const float* ya, float* outp) {
        // catb = bf16([xa | ya]); columns 0..1023 double as x_bf16 (lda=2048),
        // columns 1024..2047 as y_bf16.
        {
            const size_t total = (size_t)S * 512;
            concat_cvt_kernel<<<(unsigned)((total + 255) / 256), 256, 0, stream>>>(
                (const float4*)xa, (const float4*)ya, (u16x4*)catb, S);
        }
        // icm = softmax(cat @ W_cat + b_cat)
        run_gemm<false, false, false>(catb, D2, Wcatb, D2, lg, D2, S, D2, D2, b_cat, 1.0f, 0, stream);
        softmax_rows_bf16_kernel<<<S, 256, 0, stream>>>(lg, icmb);
        // iq = softmax(x @ W_in + b_in)
        run_gemm<false, false, false>(catb, D2, Winb, D2, lg, D2, S, D2, D, b_in, 1.0f, 0, stream);
        softmax_rows_bf16_kernel<<<S, 256, 0, stream>>>(lg, qb);
        // iqk = iq^T @ icm   (bf16 out)
        run_gemm<true, false, true>(qb, D2, icmb, D2, iqkb, D2, D2, D2, S, nullptr, 1.0f, 0, stream);
        // oq = softmax(y @ W_out + b_out)
        run_gemm<false, false, false>(catb + D, D2, Woutb, D2, lg, D2, S, D2, D, b_out, 1.0f, 0, stream);
        softmax_rows_bf16_kernel<<<S, 256, 0, stream>>>(lg, qb);
        // oqk = oq^T @ icm   (bf16 out)
        run_gemm<true, false, true>(qb, D2, icmb, D2, oqkb, D2, D2, D2, S, nullptr, 1.0f, 0, stream);
        // meta = relu(iqk @ oqk^T)   (bf16 out)
        run_gemm<false, true, true>(iqkb, D2, oqkb, D2, metab, D2, D2, D2, D2, nullptr, 1.0f, 1, stream);
        // res = (icm @ meta)/sqrt(2048) -> reuse catb (cat is dead)  (bf16 out)
        run_gemm<false, false, true>(icmb, D2, metab, D2, catb, D2, S, D2, D2, nullptr, inv_sqrt, 0, stream);
        // out = relu(res @ W_fin + b_fin)  (f32 out)
        run_gemm<false, false, false>(catb, D2, Wfinb, OUTN, outp, OUTN, S, OUTN, D2, b_fin, 1.0f, 1, stream);
    };

    march(x, yemb, out);                    // transformation
    march(x, raug, out + (size_t)S * OUTN); // random_desc
}
